// IDContextAttention_4398046511688
// MI455X (gfx1250) — matrix operand order repaired and checked by reading, not compiled, not run on hardware
//
#include <hip/hip_runtime.h>

typedef __bf16 bf16;
typedef __attribute__((ext_vector_type(16))) __bf16 v16bf;
typedef __attribute__((ext_vector_type(8)))  __bf16 v8bf;
typedef __attribute__((ext_vector_type(8)))  float   v8f;
typedef int v4i __attribute__((vector_size(16)));

constexpr int BN = 4, SEQ = 2048, HID = 768, NHEADS = 12, HDIM = 64;
constexpr int LTOT = 2049;        // S+1 keys (id anchor at index 0)
constexpr int LK   = 2112;        // K rows padded to 33*64
constexpr int LPAD = 2112;        // Vt key-stride padded to 33*64
constexpr int NKT  = LK / 64;     // 33 key tiles of 64
// HD^-0.5 * log2(e): softmax runs in base-2 (v_exp_f32 is natively 2^x)
constexpr float SCALE_LOG2 = 0.125f * 1.44269504088896340736f;

// LDS strides (in bf16 elements): 36-dword row stride -> 36N mod 64 distinct
// for N=0..15 (conflict-free b128 reads) and 16B-aligned.
constexpr int KSTR = 72;          // K tile rows (64 keys x 64 d)
constexpr int VSTR = 72;          // V tile rows (64 d x 64 keys)
constexpr int PSTR = 72;          // P staging rows (16 q x 64 keys)

// ---------------- workspace layout (bytes) ----------------
constexpr size_t SZ_XBF = (size_t)BN * SEQ * HID * 2;
constexpr size_t SZ_W   = (size_t)HID * HID * 2;
constexpr size_t OFF_XBF = 0;
constexpr size_t OFF_WQ  = OFF_XBF + SZ_XBF;
constexpr size_t OFF_WK  = OFF_WQ + SZ_W;
constexpr size_t OFF_WV  = OFF_WK + SZ_W;
constexpr size_t OFF_WO  = OFF_WV + SZ_W;
constexpr size_t OFF_Q   = OFF_WO + SZ_W;
constexpr size_t OFF_K   = OFF_Q + SZ_XBF;
constexpr size_t SZ_K    = (size_t)BN * LK * HID * 2;
constexpr size_t OFF_VT  = OFF_K + SZ_K;
constexpr size_t SZ_VT   = (size_t)BN * NHEADS * HDIM * LPAD * 2;
constexpr size_t OFF_AO  = OFF_VT + SZ_VT;

// Build a 16-element bf16 fragment from two contiguous 8-element (16B) chunks.
__device__ __forceinline__ v16bf load16(const bf16* p0, const bf16* p1) {
  v8bf lo = *(const v8bf*)p0;
  v8bf hi = *(const v8bf*)p1;
  v16bf r;
#pragma unroll
  for (int i = 0; i < 8; ++i) { r[i] = lo[i]; r[8 + i] = hi[i]; }
  return r;
}

__device__ __forceinline__ v8f wmma_bf16(v16bf a, v16bf b, v8f c) {
  return __builtin_amdgcn_wmma_f32_16x16x32_bf16(false, a, false, b, (short)0, c,
                                                 false, false);
}

__device__ __forceinline__ float fexp2(float x) {
#if __has_builtin(__builtin_amdgcn_exp2f)
  return __builtin_amdgcn_exp2f(x);   // bare v_exp_f32
#else
  return exp2f(x);
#endif
}

// ---------------- DPP16 16-lane butterfly reductions (no LDS traffic) ----------------
template <int CTRL>
__device__ __forceinline__ float dppf(float x) {
  int r = __builtin_amdgcn_update_dpp(0, __float_as_int(x), CTRL, 0xF, 0xF, true);
  return __int_as_float(r);
}
__device__ __forceinline__ float red16_max(float t) {
  t = fmaxf(t, dppf<0xB1>(t));   // quad_perm [1,0,3,2]  (xor 1)
  t = fmaxf(t, dppf<0x4E>(t));   // quad_perm [2,3,0,1]  (xor 2)
  t = fmaxf(t, dppf<0x141>(t));  // row_half_mirror       (other quad)
  t = fmaxf(t, dppf<0x140>(t));  // row_mirror            (other octet)
  return t;
}
__device__ __forceinline__ float red16_sum(float t) {
  t += dppf<0xB1>(t);
  t += dppf<0x4E>(t);
  t += dppf<0x141>(t);
  t += dppf<0x140>(t);
  return t;
}

// ---------------- async global -> LDS staging ----------------
#define AS_GLOBAL __attribute__((address_space(1)))
#define AS_LDS    __attribute__((address_space(3)))

#if __has_builtin(__builtin_amdgcn_global_load_async_to_lds_b128)
__device__ __forceinline__ void async_cp16(const bf16* g, bf16* l) {
  __builtin_amdgcn_global_load_async_to_lds_b128(
      (AS_GLOBAL v4i*)g, (AS_LDS v4i*)l, 0, 0);
}
__device__ __forceinline__ void async_wait0() {
#if __has_builtin(__builtin_amdgcn_s_wait_asynccnt)
  __builtin_amdgcn_s_wait_asynccnt(0);
#else
  asm volatile("s_wait_asynccnt 0x0" ::: "memory");
#endif
}
#else
// fallback: synchronous staging through VGPRs (still shares tiles across waves)
__device__ __forceinline__ void async_cp16(const bf16* g, bf16* l) {
  *(v8bf*)l = *(const v8bf*)g;
}
__device__ __forceinline__ void async_wait0() {}
#endif

// ---------------- f32 -> bf16 conversion ----------------
__global__ void cvt_f32_to_bf16(const float* __restrict__ in, bf16* __restrict__ out,
                                int n) {
  int i = blockIdx.x * blockDim.x + threadIdx.x;
  int stride = gridDim.x * blockDim.x;
  for (; i < n; i += stride) out[i] = (bf16)in[i];
}

// ---------------- anchor row: K_id = anc@Wk^T+bk, V_id = anc@Wv^T+bv ----------------
__global__ void anchor_kernel(const float* __restrict__ anc,
                              const float* __restrict__ Wk, const float* __restrict__ bk,
                              const float* __restrict__ Wv, const float* __restrict__ bv,
                              bf16* __restrict__ Kbf, bf16* __restrict__ Vt) {
  int tid = blockIdx.x * blockDim.x + threadIdx.x;
  if (tid >= BN * HID) return;
  int b = tid / HID, o = tid % HID;
  const float* a = anc + (size_t)b * HID;
  const float* wk = Wk + (size_t)o * HID;
  const float* wv = Wv + (size_t)o * HID;
  float ak = bk[o], av = bv[o];
  for (int h = 0; h < HID; ++h) {
    float x = a[h];
    ak += x * wk[h];
    av += x * wv[h];
  }
  Kbf[(size_t)b * LK * HID + o] = (bf16)ak;
  int head = o >> 6, d = o & 63;
  Vt[(((size_t)b * NHEADS + head) * HDIM + d) * LPAD + 0] = (bf16)av;
}

// ---------------- projection GEMM: out = A(bf16) @ W^T + bias ----------------
// Wave tile 32x32 (2 A-frags x 2 B-frags -> 4 WMMA per 8 x b128 loads).
// MODE 0: bf16 out (B,S,H), scaled by HD^-0.5*log2(e)  (Q, log2-domain softmax)
// MODE 1: bf16 out (B,LK,H), row s -> s+1              (K)
// MODE 2: bf16 out transposed (B,NH,HD,LPAD), row s -> col s+1  (V^T)
// MODE 3: f32  out (B,S,H)                             (final O projection)
template <int MODE>
__global__ void proj_kernel(const bf16* __restrict__ A, const bf16* __restrict__ W,
                            const float* __restrict__ bias, void* __restrict__ outp) {
  const int lane = threadIdx.x & 31;
  const int wave = threadIdx.x >> 5;
  const int M = lane & 15;
  const int g = lane >> 4;
  const int wr = wave >> 2, wc = wave & 3;
  const int row0 = blockIdx.y * 64 + wr * 32;
  const int col0 = blockIdx.x * 128 + wc * 32;

  v8f c00 = {}, c01 = {}, c10 = {}, c11 = {};
  const bf16* ar0 = A + (size_t)(row0 + M) * HID;
  const bf16* ar1 = A + (size_t)(row0 + 16 + M) * HID;
  const bf16* w0 = W + (size_t)(col0 + M) * HID;
  const bf16* w1 = W + (size_t)(col0 + 16 + M) * HID;

  for (int k0 = 0; k0 < HID; k0 += 32) {
    v16bf a0 = load16(ar0 + k0 + g * 8, ar0 + k0 + 16 + g * 8);
    v16bf a1 = load16(ar1 + k0 + g * 8, ar1 + k0 + 16 + g * 8);
    v16bf b0 = load16(w0 + k0 + g * 8, w0 + k0 + 16 + g * 8);
    v16bf b1 = load16(w1 + k0 + g * 8, w1 + k0 + 16 + g * 8);
    c00 = wmma_bf16(a0, b0, c00);
    c01 = wmma_bf16(a0, b1, c01);
    c10 = wmma_bf16(a1, b0, c10);
    c11 = wmma_bf16(a1, b1, c11);
  }

  const float bb0 = bias[col0 + M];
  const float bb1 = bias[col0 + 16 + M];
  const int c0 = col0 + M, c1 = col0 + 16 + M;

#pragma unroll
  for (int half = 0; half < 2; ++half) {
    v8f& a0 = half ? c10 : c00;
    v8f& a1 = half ? c11 : c01;
    int rbase = row0 + half * 16;
#pragma unroll
    for (int r = 0; r < 8; ++r) {
      int row = rbase + r + 8 * g;
      float v0 = a0[r] + bb0;
      float v1 = a1[r] + bb1;
      if (MODE == 0) { v0 *= SCALE_LOG2; v1 *= SCALE_LOG2; }
      if (MODE == 0) {
        bf16* o = (bf16*)outp;
        o[(size_t)row * HID + c0] = (bf16)v0;
        o[(size_t)row * HID + c1] = (bf16)v1;
      } else if (MODE == 1) {
        int b = row >> 11, s = row & 2047;
        bf16* o = (bf16*)outp;
        size_t base = ((size_t)b * LK + s + 1) * HID;
        o[base + c0] = (bf16)v0;
        o[base + c1] = (bf16)v1;
      } else if (MODE == 2) {
        int b = row >> 11, s = row & 2047;
        bf16* o = (bf16*)outp;
        o[(((size_t)b * NHEADS + (c0 >> 6)) * HDIM + (c0 & 63)) * LPAD + (s + 1)] = (bf16)v0;
        o[(((size_t)b * NHEADS + (c1 >> 6)) * HDIM + (c1 & 63)) * LPAD + (s + 1)] = (bf16)v1;
      } else {
        float* o = (float*)outp;
        o[(size_t)row * HID + c0] = v0;
        o[(size_t)row * HID + c1] = v1;
      }
    }
  }
}

// ---------------- flash attention ----------------
// Block: 256 threads = 8 waves, one (b,head); each wave owns 16 queries.
// 64-key tiles staged once per block into LDS (async, double-buffered);
// each thread moves two K chunks and two V chunks per tile via loop-carried
// source pointers. Per-key softmax overhead is halved vs 32-key tiles.
struct AttnShared {
  bf16 k[2][64 * KSTR];   // 64 keys x 64 d per buffer
  bf16 v[2][64 * VSTR];   // 64 d x 64 keys per buffer
  bf16 p[8][16 * PSTR];   // per-wave P repack tiles (16 q x 64 keys)
};
constexpr int KBUF = 64 * KSTR;   // elements between double-buffers
constexpr int VBUF = 64 * VSTR;

__global__ void attn_kernel(const bf16* __restrict__ Q, const bf16* __restrict__ K,
                            const bf16* __restrict__ Vt, bf16* __restrict__ AO) {
  __shared__ __align__(16) AttnShared sh;
  const int tid = threadIdx.x;
  const int lane = tid & 31;
  const int wave = tid >> 5;
  const int M = lane & 15;
  const int g = lane >> 4;
  const int b = blockIdx.x / NHEADS;
  const int h = blockIdx.x % NHEADS;
  const int q0 = blockIdx.y * 128 + wave * 16;
  const int h0 = h * HDIM;

  const bf16* Qb = Q + ((size_t)b * SEQ + q0) * HID + h0;
  const bf16* Kb = K + (size_t)b * LK * HID + h0;
  const bf16* Vb = Vt + ((size_t)b * NHEADS + h) * (size_t)HDIM * LPAD;
  bf16* sp = sh.p[wave];

  // loop-carried staging pointers: 2 K chunks + 2 V chunks (16B) per thread
  const int srow = tid >> 3, scol = (tid & 7) * 8;       // 64 rows x 8 chunks
  const bf16* ksrcA = Kb + (size_t)srow * HID + scol;
  const bf16* ksrcB = Kb + (size_t)(32 + srow) * HID + scol;
  const bf16* vsrcA = Vb + (size_t)srow * LPAD + scol;
  const bf16* vsrcB = Vb + (size_t)(32 + srow) * LPAD + scol;
  bf16* kdstA = sh.k[0] + srow * KSTR + scol;
  bf16* kdstB = kdstA + 32 * KSTR;
  bf16* vdstA = sh.v[0] + srow * VSTR + scol;
  bf16* vdstB = vdstA + 32 * VSTR;

  auto stage = [&](int buf) {
    const int kb_ = buf ? KBUF : 0, vb_ = buf ? VBUF : 0;
    async_cp16(ksrcA, kdstA + kb_);
    async_cp16(ksrcB, kdstB + kb_);
    async_cp16(vsrcA, vdstA + vb_);
    async_cp16(vsrcB, vdstB + vb_);
    ksrcA += 64 * HID;  ksrcB += 64 * HID;   // next 64-key tile
    vsrcA += 64;        vsrcB += 64;
  };

  // Q fragments (scale and log2(e) already folded in at projection time)
  v16bf aq0 = load16(Qb + M * HID + g * 8,      Qb + M * HID + 16 + g * 8);
  v16bf aq1 = load16(Qb + M * HID + 32 + g * 8, Qb + M * HID + 48 + g * 8);

  float m[8], l[8];
  v8f o0 = {}, o1 = {}, o2 = {}, o3 = {};
#pragma unroll
  for (int r = 0; r < 8; ++r) { m[r] = -1e30f; l[r] = 0.f; }

  stage(0);
  async_wait0();
  __syncthreads();

  for (int kt = 0; kt < NKT; ++kt) {
    const int cur = kt & 1;
    if (kt + 1 < NKT) stage(cur ^ 1);

    const bf16* kb = sh.k[cur];
    const bf16* vb = sh.v[cur];
    const bool masked = (kt == NKT - 1);

    // scores: 4 key-groups of 16, each = 2 WMMA over d
    v8f sc[4];
#pragma unroll
    for (int kg = 0; kg < 4; ++kg) {
      const bf16* kr = kb + (kg * 16 + M) * KSTR;
      v16bf b0 = load16(kr + g * 8, kr + 16 + g * 8);
      v16bf b1 = load16(kr + 32 + g * 8, kr + 48 + g * 8);
      v8f c = {};
      c = wmma_bf16(aq0, b0, c);
      c = wmma_bf16(aq1, b1, c);
      sc[kg] = c;
    }
    if (masked) {  // uniform branch: only the final tile pays for the tail mask
      const int key0 = kt * 64;
#pragma unroll
      for (int kg = 0; kg < 4; ++kg) {
        const bool ok = (key0 + kg * 16 + M) < LTOT;
#pragma unroll
        for (int r = 0; r < 8; ++r) sc[kg][r] = ok ? sc[kg][r] : -1e30f;
      }
    }

    // online softmax in log2 domain (one reduce pass covers 64 keys)
#pragma unroll
    for (int r = 0; r < 8; ++r) {
      float t = fmaxf(fmaxf(sc[0][r], sc[1][r]), fmaxf(sc[2][r], sc[3][r]));
      t = red16_max(t);
      float mn = fmaxf(m[r], t);
      float alpha = fexp2(m[r] - mn);
      float p0 = fexp2(sc[0][r] - mn);
      float p1 = fexp2(sc[1][r] - mn);
      float p2 = fexp2(sc[2][r] - mn);
      float p3 = fexp2(sc[3][r] - mn);
      float rs = red16_sum((p0 + p1) + (p2 + p3));
      l[r] = l[r] * alpha + rs;
      m[r] = mn;
      o0[r] *= alpha; o1[r] *= alpha; o2[r] *= alpha; o3[r] *= alpha;
      int rr = (r + 8 * g) * PSTR;
      sp[rr + M] = (bf16)p0;
      sp[rr + 16 + M] = (bf16)p1;
      sp[rr + 32 + M] = (bf16)p2;
      sp[rr + 48 + M] = (bf16)p3;
    }

    // reload P in A-matrix layout: two 16x32 fragments (keys 0..31, 32..63)
    const bf16* pr = sp + M * PSTR;
    v16bf pa0 = load16(pr + g * 8,      pr + 16 + g * 8);
    v16bf pa1 = load16(pr + 32 + g * 8, pr + 48 + g * 8);

    // PV: 4 output d-tiles x 2 key-chunks
    {
      const bf16* vp = vb + (0 + M) * VSTR;
      o0 = wmma_bf16(pa0, load16(vp + g * 8, vp + 16 + g * 8), o0);
      o0 = wmma_bf16(pa1, load16(vp + 32 + g * 8, vp + 48 + g * 8), o0);
    }
    {
      const bf16* vp = vb + (16 + M) * VSTR;
      o1 = wmma_bf16(pa0, load16(vp + g * 8, vp + 16 + g * 8), o1);
      o1 = wmma_bf16(pa1, load16(vp + 32 + g * 8, vp + 48 + g * 8), o1);
    }
    {
      const bf16* vp = vb + (32 + M) * VSTR;
      o2 = wmma_bf16(pa0, load16(vp + g * 8, vp + 16 + g * 8), o2);
      o2 = wmma_bf16(pa1, load16(vp + 32 + g * 8, vp + 48 + g * 8), o2);
    }
    {
      const bf16* vp = vb + (48 + M) * VSTR;
      o3 = wmma_bf16(pa0, load16(vp + g * 8, vp + 16 + g * 8), o3);
      o3 = wmma_bf16(pa1, load16(vp + 32 + g * 8, vp + 48 + g * 8), o3);
    }

    async_wait0();     // this wave's chunks of the next tile are in LDS
    __syncthreads();   // everyone's chunks are in; done reading current buffer
  }

  // epilogue: normalize, store bf16 to (B,S,H)
#pragma unroll
  for (int r = 0; r < 8; ++r) {
    int q = q0 + r + 8 * g;
    float inv = 1.0f / l[r];
    size_t base = ((size_t)b * SEQ + q) * HID + h0;
    AO[base + 0 + M]  = (bf16)(o0[r] * inv);
    AO[base + 16 + M] = (bf16)(o1[r] * inv);
    AO[base + 32 + M] = (bf16)(o2[r] * inv);
    AO[base + 48 + M] = (bf16)(o3[r] * inv);
  }
}

// ---------------- host launcher ----------------
extern "C" void kernel_launch(void* const* d_in, const int* in_sizes, int n_in,
                              void* d_out, int out_size, void* d_ws, size_t ws_size,
                              hipStream_t stream) {
  const float* x   = (const float*)d_in[0];
  const float* anc = (const float*)d_in[1];
  const float* Wq  = (const float*)d_in[2];
  const float* bq  = (const float*)d_in[3];
  const float* Wk  = (const float*)d_in[4];
  const float* bk  = (const float*)d_in[5];
  const float* Wv  = (const float*)d_in[6];
  const float* bv  = (const float*)d_in[7];
  const float* Wo  = (const float*)d_in[8];
  const float* bo  = (const float*)d_in[9];

  char* ws = (char*)d_ws;
  bf16* xbf = (bf16*)(ws + OFF_XBF);
  bf16* wqb = (bf16*)(ws + OFF_WQ);
  bf16* wkb = (bf16*)(ws + OFF_WK);
  bf16* wvb = (bf16*)(ws + OFF_WV);
  bf16* wob = (bf16*)(ws + OFF_WO);
  bf16* qbf = (bf16*)(ws + OFF_Q);
  bf16* kbf = (bf16*)(ws + OFF_K);
  bf16* vt  = (bf16*)(ws + OFF_VT);
  bf16* ao  = (bf16*)(ws + OFF_AO);

  const int nX = BN * SEQ * HID;
  const int nW = HID * HID;
  cvt_f32_to_bf16<<<2048, 256, 0, stream>>>(x, xbf, nX);
  cvt_f32_to_bf16<<<1024, 256, 0, stream>>>(Wq, wqb, nW);
  cvt_f32_to_bf16<<<1024, 256, 0, stream>>>(Wk, wkb, nW);
  cvt_f32_to_bf16<<<1024, 256, 0, stream>>>(Wv, wvb, nW);
  cvt_f32_to_bf16<<<1024, 256, 0, stream>>>(Wo, wob, nW);

  anchor_kernel<<<(BN * HID + 255) / 256, 256, 0, stream>>>(anc, Wk, bk, Wv, bv, kbf, vt);

  dim3 pgrid(HID / 128, (BN * SEQ) / 64);
  proj_kernel<0><<<pgrid, 256, 0, stream>>>(xbf, wqb, bq, (void*)qbf);
  proj_kernel<1><<<pgrid, 256, 0, stream>>>(xbf, wkb, bk, (void*)kbf);
  proj_kernel<2><<<pgrid, 256, 0, stream>>>(xbf, wvb, bv, (void*)vt);

  dim3 agrid(BN * NHEADS, SEQ / 128);
  attn_kernel<<<agrid, 256, 0, stream>>>(qbf, kbf, vt, ao);

  proj_kernel<3><<<pgrid, 256, 0, stream>>>(ao, wob, bo, d_out);
}
